// DeformConvWithOffset_13202729468338
// MI455X (gfx1250) — matrix operand order repaired and checked by reading, not compiled, not run on hardware
//
#include <hip/hip_runtime.h>
#include <hip/hip_bf16.h>

typedef __attribute__((ext_vector_type(16))) _Float16 v16h;
typedef __attribute__((ext_vector_type(8)))  float    v8f;

#define C_IN   128
#define CO     128
#define Hdim   64
#define Wdim   64
#define HWdim  4096
#define KKt    9
#define KD     1152          // C_IN * 9
#define NCHOFF 18
#define BATCH  8
#define BSTRIDE 40           // LDS B-tile row stride in halfs (32 + 8 pad)

union FragU {
    uint4 q[2];
    v16h  h;
};

struct Meta { int i00, i01, i10, i11; float w00, w01, w10, w11; };

// ---------------- Kernel 1: f32 -> f16 weight convert ----------------
__global__ void DCN_wconv_kernel(const float* __restrict__ w, _Float16* __restrict__ wh) {
    int i = blockIdx.x * blockDim.x + threadIdx.x;
    if (i < CO * KD) wh[i] = (_Float16)w[i];
}

// ---------------- Kernel 2: offset conv (C=128 -> 18, 3x3, pad 1) ----------------
__global__ __launch_bounds__(256)
void DCN_offconv_kernel(const float* __restrict__ x, const float* __restrict__ w_off,
                        const float* __restrict__ b_off, float* __restrict__ off) {
    __shared__ float s_w[NCHOFF * KKt];   // 162 floats, per-c slice
    const int tid = threadIdx.x;
    const int b   = blockIdx.y;
    const int p   = blockIdx.x * 256 + tid;
    const int h   = p >> 6, w = p & 63;

    int   idx9[9];
    float msk9[9];
    #pragma unroll
    for (int ky = 0; ky < 3; ++ky)
        #pragma unroll
        for (int kx = 0; kx < 3; ++kx) {
            int yy = h - 1 + ky, xx = w - 1 + kx;
            bool ok = (yy >= 0) && (yy < Hdim) && (xx >= 0) && (xx < Wdim);
            int yc = min(max(yy, 0), Hdim - 1), xc = min(max(xx, 0), Wdim - 1);
            idx9[ky * 3 + kx] = yc * Wdim + xc;
            msk9[ky * 3 + kx] = ok ? 1.f : 0.f;
        }

    float acc[NCHOFF];
    #pragma unroll
    for (int j = 0; j < NCHOFF; ++j) acc[j] = b_off[j];

    for (int c = 0; c < C_IN; ++c) {
        __syncthreads();
        if (tid < NCHOFF * KKt) {
            int j = tid / KKt, kk = tid - j * KKt;
            s_w[tid] = w_off[(size_t)j * KD + c * KKt + kk];
        }
        __syncthreads();
        const float* xp = x + (size_t)(b * C_IN + c) * HWdim;
        float xv[9];
        #pragma unroll
        for (int kk = 0; kk < 9; ++kk) xv[kk] = msk9[kk] * xp[idx9[kk]];
        #pragma unroll
        for (int j = 0; j < NCHOFF; ++j) {
            float a = acc[j];
            #pragma unroll
            for (int kk = 0; kk < 9; ++kk) a = fmaf(xv[kk], s_w[j * KKt + kk], a);
            acc[j] = a;
        }
    }
    #pragma unroll
    for (int j = 0; j < NCHOFF; ++j)
        off[(size_t)(b * NCHOFF + j) * HWdim + p] = acc[j];
}

// ---------------- Kernel 3: fused bilinear sampling + WMMA GEMM ----------------
// Block: 256 threads (8 waves), tile = Co 128 x 128 pixels, one batch.
__global__ __launch_bounds__(256)
void DCN_main_kernel(const float* __restrict__ x, const float* __restrict__ off,
                     const _Float16* __restrict__ wh, float* __restrict__ out) {
    __shared__ Meta     s_meta[KKt * 128];      // 36864 B: per-tap/per-pixel bilinear meta
    __shared__ _Float16 s_B[128 * BSTRIDE];     // 10240 B: sampled B tile [pixel][k]

    const int tid = threadIdx.x;
    const int b   = blockIdx.y;
    const int p0  = blockIdx.x * 128;

    // ---- precompute bilinear metadata (9 taps x 128 pixels) ----
    for (int e = tid; e < KKt * 128; e += 256) {
        int kk = e >> 7, pl = e & 127;
        int p = p0 + pl;
        int h = p >> 6, w = p & 63;
        int ky = kk / 3, kx = kk - ky * 3;
        float dy = off[(size_t)(b * NCHOFF + 2 * kk    ) * HWdim + p];
        float dx = off[(size_t)(b * NCHOFF + 2 * kk + 1) * HWdim + p];
        float yf = (float)(h - 1 + ky) + dy;
        float xf = (float)(w - 1 + kx) + dx;
        float fy = floorf(yf), fx = floorf(xf);
        float wy = yf - fy,   wx = xf - fx;
        int y0 = (int)fy, x0 = (int)fx;
        int y1 = y0 + 1,  x1 = x0 + 1;
        float vy0 = (y0 >= 0 && y0 < Hdim) ? 1.f : 0.f;
        float vy1 = (y1 >= 0 && y1 < Hdim) ? 1.f : 0.f;
        float vx0 = (x0 >= 0 && x0 < Wdim) ? 1.f : 0.f;
        float vx1 = (x1 >= 0 && x1 < Wdim) ? 1.f : 0.f;
        int y0c = min(max(y0, 0), Hdim - 1), y1c = min(max(y1, 0), Hdim - 1);
        int x0c = min(max(x0, 0), Wdim - 1), x1c = min(max(x1, 0), Wdim - 1);
        Meta m;
        m.i00 = y0c * Wdim + x0c; m.i01 = y0c * Wdim + x1c;
        m.i10 = y1c * Wdim + x0c; m.i11 = y1c * Wdim + x1c;
        m.w00 = (1.f - wy) * (1.f - wx) * vy0 * vx0;
        m.w01 = (1.f - wy) * wx         * vy0 * vx1;
        m.w10 = wy         * (1.f - wx) * vy1 * vx0;
        m.w11 = wy         * wx         * vy1 * vx1;
        s_meta[e] = m;
    }
    __syncthreads();

    const int lane   = tid & 31;
    const int wv     = tid >> 5;
    const int m_base = (wv & 3) * 32;   // 4 waves along M (Co)
    const int n_base = (wv >> 2) * 64;  // 2 waves along N (pixels)
    const int lrow   = lane & 15;
    const int khalf  = lane >> 4;

    v8f zero = {};
    v8f acc[2][4];
    #pragma unroll
    for (int i = 0; i < 2; ++i)
        #pragma unroll
        for (int j = 0; j < 4; ++j) acc[i][j] = zero;

    const int pl_s = tid >> 1;          // sampler: pixel 0..127
    const int ks_s = (tid & 1) * 16;    // sampler: k sub-range within chunk
    const float* xb_base = x + (size_t)b * C_IN * HWdim;

    for (int kc = 0; kc < KD / 32; ++kc) {
        // ---- stage sampled f16 B tile into LDS: s_B[p][k], 16 halfs/thread ----
        {
            const int k0 = kc * 32 + ks_s;
            union { _Float16 hh[16]; uint4 q[2]; } pk;
            #pragma unroll
            for (int i = 0; i < 16; ++i) {
                int k  = k0 + i;
                int c  = k / 9;
                int kk = k - c * 9;
                const Meta m = s_meta[kk * 128 + pl_s];
                const float* xp = xb_base + (size_t)c * HWdim;
                float v = m.w00 * xp[m.i00] + m.w01 * xp[m.i01]
                        + m.w10 * xp[m.i10] + m.w11 * xp[m.i11];
                pk.hh[i] = (_Float16)v;
            }
            uint4* dst = (uint4*)&s_B[pl_s * BSTRIDE + ks_s];
            dst[0] = pk.q[0];
            dst[1] = pk.q[1];
        }
        __syncthreads();

        // ---- A fragments: 16-bit A 16x32 layout (K = khalf*8 and 16+khalf*8) ----
        FragU a0, a1;
        {
            const _Float16* pa = wh + (size_t)(m_base + lrow) * KD + kc * 32 + khalf * 8;
            a0.q[0] = *(const uint4*)(pa);
            a0.q[1] = *(const uint4*)(pa + 16);
            const _Float16* pb = wh + (size_t)(m_base + 16 + lrow) * KD + kc * 32 + khalf * 8;
            a1.q[0] = *(const uint4*)(pb);
            a1.q[1] = *(const uint4*)(pb + 16);
        }

        // ---- B fragments from LDS (K = khalf*8 and 16+khalf*8) + WMMA ----
        #pragma unroll
        for (int nt = 0; nt < 4; ++nt) {
            FragU bf;
            const uint4* src = (const uint4*)&s_B[(n_base + nt * 16 + lrow) * BSTRIDE + khalf * 8];
            bf.q[0] = src[0];
            bf.q[1] = src[2];
            acc[0][nt] = __builtin_amdgcn_wmma_f32_16x16x32_f16(
                false, a0.h, false, bf.h, (short)0, acc[0][nt], false, false);
            acc[1][nt] = __builtin_amdgcn_wmma_f32_16x16x32_f16(
                false, a1.h, false, bf.h, (short)0, acc[1][nt], false, false);
        }
        __syncthreads();
    }

    // ---- writeback: C/D layout (lane<16 -> M=r, lane>=16 -> M=r+8) ----
    #pragma unroll
    for (int mt = 0; mt < 2; ++mt)
        #pragma unroll
        for (int nt = 0; nt < 4; ++nt) {
            int n = p0 + n_base + nt * 16 + lrow;
            #pragma unroll
            for (int r = 0; r < 8; ++r) {
                int mrow = m_base + mt * 16 + khalf * 8 + r;
                out[(size_t)(b * CO + mrow) * HWdim + n] = acc[mt][nt][r];
            }
        }
}

extern "C" void kernel_launch(void* const* d_in, const int* in_sizes, int n_in,
                              void* d_out, int out_size, void* d_ws, size_t ws_size,
                              hipStream_t stream) {
    const float* x     = (const float*)d_in[0];   // (8,128,64,64)
    const float* w_off = (const float*)d_in[1];   // (18,128,3,3)
    const float* b_off = (const float*)d_in[2];   // (18,)
    const float* w_dcn = (const float*)d_in[3];   // (128,128,3,3)
    float* out = (float*)d_out;                   // (8,128,64,64)

    // Workspace: [off f32: 8*18*4096] [wh f16: 128*1152]
    float*    off_ws = (float*)d_ws;
    _Float16* wh     = (_Float16*)((char*)d_ws + (size_t)BATCH * NCHOFF * HWdim * sizeof(float));

    DCN_wconv_kernel<<<(CO * KD + 255) / 256, 256, 0, stream>>>(w_dcn, wh);
    DCN_offconv_kernel<<<dim3(HWdim / 256, BATCH), 256, 0, stream>>>(x, w_off, b_off, off_ws);
    DCN_main_kernel<<<dim3(HWdim / 128, BATCH), 256, 0, stream>>>(x, off_ws, wh, out);
}
